// Preprocessing2_32014686224841
// MI455X (gfx1250) — compile-verified, matrix operand order checked by reading, not run on hardware
//
#include <hip/hip_runtime.h>
#include <hip/hip_fp16.h>
#include <math.h>

typedef __attribute__((ext_vector_type(16))) _Float16 v16h;
typedef __attribute__((ext_vector_type(8)))  float    v8f;
typedef __attribute__((ext_vector_type(4)))  float    f4;

#define BATCH 32
#define CH    3
#define HH    640
#define WW    640
#define HID   64
#define KF    (CH * 64)          // 192 features
#define NP    (CH * CH + 2 * CH) // 15 params per sample

// ---------------------------------------------------------------------------
// Kernel 1: 8x8 adaptive average pool.
// grid = (96 planes, 8 pool-rows), block = 640 threads (one per column).
// Default RT loads on purpose: this pass installs x into the 192 MB L2 so the
// apply pass can re-read it without HBM traffic.
// ---------------------------------------------------------------------------
__global__ void pool_kernel(const float* __restrict__ x, float* __restrict__ feat) {
    __shared__ float cell[8];
    const int tid   = threadIdx.x;    // column 0..639
    const int plane = blockIdx.x;     // b*3 + c, 0..95
    const int py    = blockIdx.y;     // pool row 0..7
    if (tid < 8) cell[tid] = 0.0f;
    __syncthreads();

    const float* base = x + ((size_t)plane * HH + (size_t)py * 80) * WW + tid;
    float s = 0.0f;
#pragma unroll 8
    for (int r = 0; r < 80; ++r) s += base[(size_t)r * WW];
    atomicAdd(&cell[tid / 80], s);    // ds_add_f32
    __syncthreads();

    if (tid < 8) {
        const int b = plane / CH, c = plane % CH;
        // torch .view flatten order: feat[b][c*64 + py*8 + px]
        feat[(size_t)b * KF + c * 64 + py * 8 + tid] = cell[tid] * (1.0f / 6400.0f);
    }
}

// ---------------------------------------------------------------------------
// Kernel 2: per-sample MLP on a single wave32 using V_WMMA_F32_16X16X32_F16.
//   hid   = clip(feat @ W1 + b1, 0, 6)      (32x192 @ 192x64)
//   param = sigmoid(hid @ W2)               (32x64  @ 64x15, N padded to 16)
// Fragment packing per ISA 7.12.2 (16-bit A 16x32): lane l holds row m=l&15;
// half element i maps to k_local = (i/8)*16 + (l>>4)*8 + (i%8). B mirrored.
// ---------------------------------------------------------------------------
__global__ void mlp_kernel(const float* __restrict__ feat,
                           const float* __restrict__ W1,
                           const float* __restrict__ b1,
                           const float* __restrict__ W2,
                           float* __restrict__ params) {
    __shared__ float hid[BATCH][HID];
    const int l  = threadIdx.x;  // lane 0..31, EXEC all ones
    const int lo = l & 15;
    const int hi = l >> 4;

    // ---- GEMM1: 2 M-tiles x 4 N-tiles x 6 K-steps of 32 ----
    for (int mt = 0; mt < 2; ++mt) {
        for (int nt = 0; nt < 4; ++nt) {
            v8f acc = {};
            for (int kt = 0; kt < KF; kt += 32) {
                v16h a, b;
#pragma unroll
                for (int i = 0; i < 16; ++i) {
                    const int kl = ((i >> 3) << 4) + hi * 8 + (i & 7);
                    a[i] = (_Float16)feat[(size_t)(mt * 16 + lo) * KF + kt + kl];
                    b[i] = (_Float16)W1[(size_t)(kt + kl) * HID + nt * 16 + lo];
                }
                acc = __builtin_amdgcn_wmma_f32_16x16x32_f16(
                    false, a, false, b, (short)0, acc, false, false);
            }
#pragma unroll
            for (int r = 0; r < 8; ++r) {
                const int m = mt * 16 + hi * 8 + r;
                const int n = nt * 16 + lo;
                float v = acc[r] + b1[n];
                v = fminf(fmaxf(v, 0.0f), 6.0f);   // ReLU6
                hid[m][n] = v;
            }
        }
    }
    __syncthreads();

    // ---- GEMM2: 2 M-tiles x 1 N-tile (15->16) x 2 K-steps of 32 ----
    for (int mt = 0; mt < 2; ++mt) {
        v8f acc = {};
        for (int kt = 0; kt < HID; kt += 32) {
            v16h a, b;
#pragma unroll
            for (int i = 0; i < 16; ++i) {
                const int kl = ((i >> 3) << 4) + hi * 8 + (i & 7);
                a[i] = (_Float16)hid[mt * 16 + lo][kt + kl];
                b[i] = (lo < NP) ? (_Float16)W2[(size_t)(kt + kl) * NP + lo]
                                 : (_Float16)0.0f;    // v_cndmask pad, no EXEC change
            }
            acc = __builtin_amdgcn_wmma_f32_16x16x32_f16(
                false, a, false, b, (short)0, acc, false, false);
        }
#pragma unroll
        for (int r = 0; r < 8; ++r) {
            const int m = mt * 16 + hi * 8 + r;
            if (lo < NP)
                params[(size_t)m * NP + lo] = 1.0f / (1.0f + __expf(-acc[r]));
        }
    }
}

// ---------------------------------------------------------------------------
// Kernel 3: bandwidth-bound apply pass. B128 loads/stores, all three channels
// per pixel per thread. out = ((M.rgb + bias)^gamma - MEAN)/STD.
// Non-temporal loads (last use of x) and non-temporal streaming stores (out is
// never re-read) keep the pass from thrashing the 192 MB L2.
// grid = (HW/4/256, 32), block = 256 (8 waves).
// ---------------------------------------------------------------------------
__global__ void apply_kernel(const float* __restrict__ x,
                             const float* __restrict__ params,
                             float* __restrict__ out) {
    __shared__ float p[NP];
    const int b = blockIdx.y;
    const int t = threadIdx.x;
    if (t < NP) p[t] = params[(size_t)b * NP + t];
    __syncthreads();

    const size_t HW4  = (size_t)HH * WW / 4;  // 102400 float4 per plane
    const size_t idx4 = (size_t)blockIdx.x * blockDim.x + t;

    const f4* xb = (const f4*)(x + (size_t)b * CH * HH * WW);
    f4*       ob = (f4*)(out + (size_t)b * CH * HH * WW);

    const f4 r  = __builtin_nontemporal_load(xb + idx4);
    const f4 g  = __builtin_nontemporal_load(xb + idx4 + HW4);
    const f4 bl = __builtin_nontemporal_load(xb + idx4 + 2 * HW4);

    const float mean[3] = {0.485f, 0.456f, 0.406f};
    const float stdv[3] = {0.229f, 0.224f, 0.225f};

#pragma unroll
    for (int i = 0; i < 3; ++i) {
        const float m0 = p[i * 3 + 0], m1 = p[i * 3 + 1], m2 = p[i * 3 + 2];
        const float bias = p[9 + i], gam = p[12 + i];
        const float inv = 1.0f / stdv[i];
        f4 v;
        v.x = (__powf(m0 * r.x + m1 * g.x + m2 * bl.x + bias, gam) - mean[i]) * inv;
        v.y = (__powf(m0 * r.y + m1 * g.y + m2 * bl.y + bias, gam) - mean[i]) * inv;
        v.z = (__powf(m0 * r.z + m1 * g.z + m2 * bl.z + bias, gam) - mean[i]) * inv;
        v.w = (__powf(m0 * r.w + m1 * g.w + m2 * bl.w + bias, gam) - mean[i]) * inv;
        __builtin_nontemporal_store(v, ob + idx4 + (size_t)i * HW4);
    }
}

// ---------------------------------------------------------------------------
extern "C" void kernel_launch(void* const* d_in, const int* in_sizes, int n_in,
                              void* d_out, int out_size, void* d_ws, size_t ws_size,
                              hipStream_t stream) {
    const float* x  = (const float*)d_in[0];  // (32,3,640,640)
    const float* W1 = (const float*)d_in[1];  // (192,64)
    const float* b1 = (const float*)d_in[2];  // (64,)
    const float* W2 = (const float*)d_in[3];  // (64,15)
    float* out = (float*)d_out;

    float* feat   = (float*)d_ws;                         // 32*192 floats
    float* params = (float*)d_ws + (size_t)BATCH * KF;    // 32*15 floats

    // Pass 1: pooled features (reads 157 MB, installs x into L2)
    pool_kernel<<<dim3(BATCH * CH, 8), 640, 0, stream>>>(x, feat);

    // Pass 2: tiny WMMA MLP -> per-sample color params (negligible time)
    mlp_kernel<<<1, 32, 0, stream>>>(feat, W1, b1, W2, params);

    // Pass 3: per-pixel color transform (re-reads x — likely L2-resident —
    // streams 157 MB of output with NT stores)
    const int hw4 = HH * WW / 4;                          // 102400
    apply_kernel<<<dim3(hw4 / 256, BATCH), 256, 0, stream>>>(x, params, out);
}
